// Attention_48670569398699
// MI455X (gfx1250) — matrix operand order repaired and checked by reading, not compiled, not run on hardware
//
#include <hip/hip_runtime.h>

#define BATCH  4
#define SEQ    2048
#define DMODEL 1024
#define NH     16
#define DHEAD  64

typedef __bf16 bf16_t;
typedef __attribute__((ext_vector_type(16))) __bf16 v16bf;
typedef __attribute__((ext_vector_type(8)))  float  v8f;

union FragU {
    uint4 u[2];
    v16bf v;
};

__device__ inline bf16_t f2bf(float f) {
    unsigned int u = __float_as_uint(f);
    unsigned int r = (u + 0x7FFFu + ((u >> 16) & 1u)) >> 16;  // round-to-nearest-even
    unsigned short hb = (unsigned short)r;
    bf16_t b;
    __builtin_memcpy(&b, &hb, 2);
    return b;
}

// CDNA5 async global->LDS copy (ASYNCcnt-tracked). VDST carries the LDS byte
// offset; low 32 bits of a generic shared pointer ARE the LDS offset (ISA
// 10.2 aperture mapping).
__device__ inline void async_copy_b128(const bf16_t* lds_dst, const bf16_t* gsrc) {
    unsigned lds_off = (unsigned)(uintptr_t)lds_dst;
    unsigned long long ga = (unsigned long long)(uintptr_t)gsrc;
    asm volatile("global_load_async_to_lds_b128 %0, %1, off"
                 :: "v"(lds_off), "v"(ga) : "memory");
}

__device__ inline void wait_async0() {
    asm volatile("s_wait_asynccnt 0x0" ::: "memory");
}

// ---------------------------------------------------------------------------
// fp32 -> bf16 cast of x and the four weight matrices
// ---------------------------------------------------------------------------
__global__ __launch_bounds__(256) void cast_all(
    const float* __restrict__ x,
    const float* __restrict__ wq, const float* __restrict__ wk,
    const float* __restrict__ wv, const float* __restrict__ wo,
    bf16_t* __restrict__ xb,
    bf16_t* __restrict__ wqb, bf16_t* __restrict__ wkb,
    bf16_t* __restrict__ wvb, bf16_t* __restrict__ wob)
{
    const int XN = BATCH * SEQ * DMODEL;
    const int WN = DMODEL * DMODEL;
    const int total = XN + 4 * WN;
    for (int i = blockIdx.x * blockDim.x + threadIdx.x; i < total;
         i += gridDim.x * blockDim.x) {
        if (i < XN)               xb[i]                = f2bf(x[i]);
        else if (i < XN + WN)     wqb[i - XN]          = f2bf(wq[i - XN]);
        else if (i < XN + 2 * WN) wkb[i - XN - WN]     = f2bf(wk[i - XN - WN]);
        else if (i < XN + 3 * WN) wvb[i - XN - 2 * WN] = f2bf(wv[i - XN - 2 * WN]);
        else                      wob[i - XN - 3 * WN] = f2bf(wo[i - XN - 3 * WN]);
    }
}

// ---------------------------------------------------------------------------
// bf16 GEMM: C[M,N] = alpha * A[M,K] @ B[K,N], fp32 accumulation (WMMA bf16).
// 128x128 block tile, 32-deep K slices, DOUBLE-BUFFERED LDS staged with
// global_load_async_to_lds_b128: tile k+1 streams in (ASYNCcnt) while the
// 8 waves run WMMAs on tile k.
// mode 0: store bf16 row-major           (Q with alpha=0.125, V)
// mode 1: store bf16 as per-head K^T     kt[(b*NH+h)*DHEAD+d][SEQ] layout
// mode 2: store fp32 row-major           (final output projection)
// ---------------------------------------------------------------------------
__global__ __launch_bounds__(256) void gemm_bf16(
    const bf16_t* __restrict__ A, const bf16_t* __restrict__ Bw,
    bf16_t* __restrict__ Cb, float* __restrict__ Cf,
    int M, int N, int K, float alpha, int mode)
{
    __shared__ bf16_t As[2][128 * 32];   // 2 x 8 KB
    __shared__ bf16_t Bs[2][32 * 128];   // 2 x 8 KB

    const int tid  = threadIdx.x;
    const int lane = tid & 31;
    const int wid  = tid >> 5;      // 0..7
    const int wm   = wid >> 2;      // 0..1
    const int wn   = wid & 3;       // 0..3
    const int g    = lane >> 4;     // lane group (C-layout row offset 8*g)
    const int ln   = lane & 15;

    const int bm = blockIdx.y * 128;
    const int bn = blockIdx.x * 128;

    // stage one 128x32 A tile + 32x128 B tile into LDS buffer `buf`
    auto stage = [&](int ktEl, int buf) {
        #pragma unroll
        for (int i = 0; i < 2; i++) {
            int c  = tid + i * 256;
            int ar = c >> 2, ac = (c & 3) * 8;
            async_copy_b128(&As[buf][ar * 32 + ac],
                            &A[(bm + ar) * K + ktEl + ac]);
            int br = c >> 4, bc = (c & 15) * 8;
            async_copy_b128(&Bs[buf][br * 128 + bc],
                            &Bw[(ktEl + br) * N + bn + bc]);
        }
    };

    const v8f vzero = {0.f, 0.f, 0.f, 0.f, 0.f, 0.f, 0.f, 0.f};
    v8f acc[4][2];
    for (int i = 0; i < 4; i++)
        for (int j = 0; j < 2; j++)
            acc[i][j] = vzero;

    const int nk = K / 32;
    stage(0, 0);
    wait_async0();
    __syncthreads();

    for (int kt = 0; kt < nk; kt++) {
        const int buf = kt & 1;
        if (kt + 1 < nk) stage((kt + 1) * 32, buf ^ 1);

        // A fragments: lane ln holds row (..+ln), K = g*8..g*8+7 and +16.
        FragU af[4], bf[2];
        #pragma unroll
        for (int fm = 0; fm < 4; fm++) {
            const bf16_t* p = &As[buf][(wm * 64 + fm * 16 + ln) * 32 + g * 8];
            af[fm].u[0] = *(const uint4*)(p);
            af[fm].u[1] = *(const uint4*)(p + 16);
        }
        // B fragments: lane ln holds column N (..+ln), K = g*8..g*8+7 and +16.
        #pragma unroll
        for (int fn = 0; fn < 2; fn++) {
            #pragma unroll
            for (int s = 0; s < 16; s++)
                bf[fn].v[s] = Bs[buf][(g * 8 + (s & 7) + (s >> 3) * 16) * 128 + wn * 32 + fn * 16 + ln];
        }
        #pragma unroll
        for (int fm = 0; fm < 4; fm++)
            #pragma unroll
            for (int fn = 0; fn < 2; fn++)
                acc[fm][fn] = __builtin_amdgcn_wmma_f32_16x16x32_bf16(
                    false, af[fm].v, false, bf[fn].v,
                    (short)0, acc[fm][fn], false, false);

        wait_async0();     // next tile landed in the other buffer
        __syncthreads();   // everyone done reading buf; writes visible
    }

    // Epilogue. C element (r, lane) -> (M = r + 8*g, N = ln).
    #pragma unroll
    for (int fm = 0; fm < 4; fm++) {
        #pragma unroll
        for (int fn = 0; fn < 2; fn++) {
            #pragma unroll
            for (int r = 0; r < 8; r++) {
                int gm = bm + wm * 64 + fm * 16 + r + 8 * g;
                int gn = bn + wn * 32 + fn * 16 + ln;
                float val = acc[fm][fn][r] * alpha;
                if (mode == 0) {
                    Cb[gm * N + gn] = f2bf(val);
                } else if (mode == 1) {
                    int hh = gn >> 6, dd = gn & 63;
                    int bb = gm >> 11, ss = gm & 2047;
                    Cb[((bb * NH + hh) * DHEAD + dd) * SEQ + ss] = f2bf(val);
                } else {
                    Cf[gm * N + gn] = val;
                }
            }
        }
    }
}

// ---------------------------------------------------------------------------
// Flash attention: grid (SEQ/128, BATCH*NH), 256 threads = 8 waves.
// Each wave owns 16 query rows of one (b,h). Online softmax over 64-wide
// KV tiles. K^T and V tiles are staged once per BLOCK into double-buffered
// LDS via async copies (8x less global traffic than per-wave loads); Q
// A-frags stay resident in registers. P is re-laid-out C-frag -> A-frag
// through a per-wave LDS buffer (same-wave LDS ops are in-order).
// ---------------------------------------------------------------------------
__global__ __launch_bounds__(256) void flash_attn(
    const bf16_t* __restrict__ qb, const bf16_t* __restrict__ ktb,
    const bf16_t* __restrict__ vb, const float* __restrict__ bias,
    bf16_t* __restrict__ ab)
{
    __shared__ bf16_t Kts[2][64 * 64];    // 2 x 8 KB, [d][kv]
    __shared__ bf16_t Vs[2][64 * 64];     // 2 x 8 KB, [kv][d]
    __shared__ bf16_t Pbuf[8][16 * 64];   // 16 KB, one 16x64 P tile per wave

    const int tid  = threadIdx.x;
    const int lane = tid & 31;
    const int w    = tid >> 5;
    const int g    = lane >> 4;
    const int ln   = lane & 15;

    const int bh = blockIdx.y;
    const int b  = bh >> 4;
    const int h  = bh & 15;
    const int q0 = blockIdx.x * 128 + w * 16;

    const bf16_t* ktbase = ktb + (size_t)bh * DHEAD * SEQ;
    const bf16_t* vbase  = vb + (size_t)(b * SEQ) * DMODEL + h * DHEAD;
    const float*  biasb  = bias + b * SEQ;

    // stage 64x64 K^T tile ([d][kv]) and 64x64 V tile ([kv][d]) for kv0
    auto stage = [&](int kv0, int buf) {
        #pragma unroll
        for (int i = 0; i < 2; i++) {
            int c   = tid + i * 256;
            int row = c >> 3, cc = (c & 7) * 8;
            async_copy_b128(&Kts[buf][row * 64 + cc],
                            ktbase + (size_t)row * SEQ + kv0 + cc);
            async_copy_b128(&Vs[buf][row * 64 + cc],
                            vbase + (size_t)(kv0 + row) * DMODEL + cc);
        }
    };

    // Resident Q A-fragments (depth halves 0..31, 32..63), already *0.125.
    FragU qa[2];
    {
        const bf16_t* qrow = qb + (size_t)(b * SEQ + q0 + ln) * DMODEL + h * DHEAD;
        #pragma unroll
        for (int kf = 0; kf < 2; kf++) {
            qa[kf].u[0] = *(const uint4*)(qrow + kf * 32 + g * 8);
            qa[kf].u[1] = *(const uint4*)(qrow + kf * 32 + g * 8 + 16);
        }
    }

    const v8f vzero = {0.f, 0.f, 0.f, 0.f, 0.f, 0.f, 0.f, 0.f};
    v8f o[4];
    for (int i = 0; i < 4; i++) o[i] = vzero;
    float mrow[8], lrow[8];
    for (int r = 0; r < 8; r++) { mrow[r] = -1e30f; lrow[r] = 0.0f; }

    const int niter = SEQ / 64;
    stage(0, 0);
    wait_async0();
    __syncthreads();

    for (int it = 0; it < niter; it++) {
        const int kv0 = it * 64;
        const int buf = it & 1;
        if (it + 1 < niter) stage((it + 1) * 64, buf ^ 1);

        // ---- logits: S[16q x 64kv] = Q(16x64) @ K^T(64x64kv) ----
        v8f sfr[4];
        #pragma unroll
        for (int nf = 0; nf < 4; nf++) {
            sfr[nf] = vzero;
            #pragma unroll
            for (int kh = 0; kh < 2; kh++) {
                FragU kf_;
                #pragma unroll
                for (int s = 0; s < 16; s++)
                    kf_.v[s] = Kts[buf][(kh * 32 + g * 8 + (s & 7) + (s >> 3) * 16) * 64 + nf * 16 + ln];
                sfr[nf] = __builtin_amdgcn_wmma_f32_16x16x32_bf16(
                    false, qa[kh].v, false, kf_.v, (short)0, sfr[nf],
                    false, false);
            }
            float bv = biasb[kv0 + nf * 16 + ln];
            #pragma unroll
            for (int r = 0; r < 8; r++) sfr[nf][r] += bv;
        }

        // ---- online softmax: row max over this tile (reduce over N) ----
        float tmax[8];
        #pragma unroll
        for (int r = 0; r < 8; r++)
            tmax[r] = fmaxf(fmaxf(sfr[0][r], sfr[1][r]),
                            fmaxf(sfr[2][r], sfr[3][r]));
        #pragma unroll
        for (int mask = 1; mask < 16; mask <<= 1)
            #pragma unroll
            for (int r = 0; r < 8; r++)
                tmax[r] = fmaxf(tmax[r], __shfl_xor(tmax[r], mask, 32));

        float mnew[8], corr[8], ls[8];
        #pragma unroll
        for (int r = 0; r < 8; r++) {
            mnew[r] = fmaxf(mrow[r], tmax[r]);
            corr[r] = __expf(mrow[r] - mnew[r]);
            ls[r]   = 0.0f;
        }

        // ---- P = exp(S - mnew): accumulate row sums, stage bf16 into LDS ----
        #pragma unroll
        for (int nf = 0; nf < 4; nf++)
            #pragma unroll
            for (int r = 0; r < 8; r++) {
                float p = __expf(sfr[nf][r] - mnew[r]);
                ls[r] += p;
                Pbuf[w][(r + 8 * g) * 64 + nf * 16 + ln] = f2bf(p);
            }

        #pragma unroll
        for (int mask = 1; mask < 16; mask <<= 1)
            #pragma unroll
            for (int r = 0; r < 8; r++)
                ls[r] += __shfl_xor(ls[r], mask, 32);

        #pragma unroll
        for (int r = 0; r < 8; r++) {
            lrow[r] = lrow[r] * corr[r] + ls[r];
            mrow[r] = mnew[r];
        }
        #pragma unroll
        for (int df = 0; df < 4; df++)
            #pragma unroll
            for (int r = 0; r < 8; r++)
                o[df][r] *= corr[r];

        // ---- P in A-fragment layout (same-wave LDS RAW is in-order) ----
        FragU pa[2];
        #pragma unroll
        for (int pf = 0; pf < 2; pf++) {
            const bf16_t* p = &Pbuf[w][ln * 64 + pf * 32 + g * 8];
            pa[pf].u[0] = *(const uint4*)(p);
            pa[pf].u[1] = *(const uint4*)(p + 16);
        }

        // ---- O += P(16x64kv) @ V(64kv x 64d) ----
        #pragma unroll
        for (int df = 0; df < 4; df++) {
            #pragma unroll
            for (int ph = 0; ph < 2; ph++) {
                FragU vf;
                #pragma unroll
                for (int s = 0; s < 16; s++)
                    vf.v[s] = Vs[buf][(ph * 32 + g * 8 + (s & 7) + (s >> 3) * 16) * 64 + df * 16 + ln];
                o[df] = __builtin_amdgcn_wmma_f32_16x16x32_bf16(
                    false, pa[ph].v, false, vf.v, (short)0, o[df],
                    false, false);
            }
        }

        wait_async0();     // next K^T/V tiles landed in the other buffer
        __syncthreads();
    }

    // ---- normalize and store attn output (bf16, [B*S, D] layout) ----
    #pragma unroll
    for (int df = 0; df < 4; df++)
        #pragma unroll
        for (int r = 0; r < 8; r++) {
            float val = o[df][r] / lrow[r];
            int row = b * SEQ + q0 + r + 8 * g;
            int col = h * DHEAD + df * 16 + ln;
            ab[(size_t)row * DMODEL + col] = f2bf(val);
        }
}

// ---------------------------------------------------------------------------
// Host-side launcher
// ---------------------------------------------------------------------------
extern "C" void kernel_launch(void* const* d_in, const int* in_sizes, int n_in,
                              void* d_out, int out_size, void* d_ws, size_t ws_size,
                              hipStream_t stream) {
    const float* x    = (const float*)d_in[0];
    const float* bias = (const float*)d_in[1];
    const float* Wq   = (const float*)d_in[2];
    const float* Wk   = (const float*)d_in[3];
    const float* Wv   = (const float*)d_in[4];
    const float* Wo   = (const float*)d_in[5];
    float* out = (float*)d_out;

    char* ws = (char*)d_ws;
    const size_t XB = (size_t)BATCH * SEQ * DMODEL * sizeof(bf16_t); // 16 MB
    const size_t WB = (size_t)DMODEL * DMODEL * sizeof(bf16_t);      // 2 MB
    bf16_t* xb  = (bf16_t*)(ws);
    bf16_t* wqb = (bf16_t*)(ws + XB);
    bf16_t* wkb = (bf16_t*)(ws + XB + 1 * WB);
    bf16_t* wvb = (bf16_t*)(ws + XB + 2 * WB);
    bf16_t* wob = (bf16_t*)(ws + XB + 3 * WB);
    bf16_t* qb  = (bf16_t*)(ws + 1 * XB + 4 * WB);
    bf16_t* ktb = (bf16_t*)(ws + 2 * XB + 4 * WB);
    bf16_t* vb  = (bf16_t*)(ws + 3 * XB + 4 * WB);
    bf16_t* ab  = (bf16_t*)(ws + 4 * XB + 4 * WB);

    cast_all<<<2048, 256, 0, stream>>>(x, Wq, Wk, Wv, Wo,
                                       xb, wqb, wkb, wvb, wob);

    dim3 gg(DMODEL / 128, (BATCH * SEQ) / 128);
    // Q projection, scale 1/sqrt(DHEAD) folded in.
    gemm_bf16<<<gg, 256, 0, stream>>>(xb, wqb, qb, nullptr,
                                      BATCH * SEQ, DMODEL, DMODEL, 0.125f, 0);
    // K projection, stored transposed per head: kt[(b*NH+h)*64+d][SEQ].
    gemm_bf16<<<gg, 256, 0, stream>>>(xb, wkb, ktb, nullptr,
                                      BATCH * SEQ, DMODEL, DMODEL, 1.0f, 1);
    // V projection, natural layout.
    gemm_bf16<<<gg, 256, 0, stream>>>(xb, wvb, vb, nullptr,
                                      BATCH * SEQ, DMODEL, DMODEL, 1.0f, 0);

    flash_attn<<<dim3(SEQ / 128, BATCH * NH), 256, 0, stream>>>(
        qb, ktb, vb, bias, ab);

    // Output projection in fp32.
    gemm_bf16<<<gg, 256, 0, stream>>>(ab, wob, nullptr, out,
                                      BATCH * SEQ, DMODEL, DMODEL, 1.0f, 2);
}
